// ALSTM_88570815578423
// MI455X (gfx1250) — matrix operand order repaired and checked by reading, not compiled, not run on hardware
//
#include <hip/hip_runtime.h>
#include <hip/hip_bf16.h>

#define B_    32
#define T_    512
#define IN_   50
#define H_    512
#define OUT_  66
#define NPOND 10
#define NWG   32          // 2 batch groups x 16 WGPs
#define HSTR  520         // LDS row stride in bf16 (1040B = 65 dword-quads, odd -> conflict-free b128)

typedef __bf16 bf16;
typedef __attribute__((ext_vector_type(16))) __bf16 v16bf;
typedef __attribute__((ext_vector_type(8)))  float  v8f;
typedef int v4i __attribute__((ext_vector_type(4)));
typedef v4i __attribute__((address_space(1))) v4i_glob;   // global (AS1) int4
typedef v4i __attribute__((address_space(3))) v4i_lds;    // LDS (AS3) int4

// Async global->LDS path (CDNA5): use if this toolchain exposes the builtins.
#if defined(__has_builtin)
#if __has_builtin(__builtin_amdgcn_global_load_async_to_lds_b128) && \
    __has_builtin(__builtin_amdgcn_s_wait_asynccnt)
#define USE_ASYNC_LDS 1
#endif
#endif

__device__ __forceinline__ bf16 f2bf(float f) {
  unsigned u = __builtin_bit_cast(unsigned, f);
  u += 0x7fffu + ((u >> 16) & 1u);                  // round-to-nearest-even
  unsigned short hs = (unsigned short)(u >> 16);
  return __builtin_bit_cast(bf16, hs);
}
__device__ __forceinline__ float bf2f(bf16 h) {
  unsigned short hs = __builtin_bit_cast(unsigned short, h);
  unsigned u = ((unsigned)hs) << 16;
  return __builtin_bit_cast(float, u);
}
__device__ __forceinline__ float sigf(float x) { return 1.0f / (1.0f + __expf(-x)); }

// ---------------- shared memory layout (dynamic LDS, ~269 KB < 320 KB/WGP) ----
struct Smem {
  bf16  W[128 * HSTR];       // W_hh slice: 128 gate rows x 512 (rows 16B aligned)
  bf16  Wout[80 * HSTR];     // W_out padded to 80 rows (only WG lid==0 uses it)
  bf16  h[16 * HSTR];        // current full h for this batch group (16 x 512)
  bf16  Wih[128 * 64];       // W_ih slice (K padded 51->64, flag col handled separately)
  bf16  xt[16 * 64];         // x_t tile, padded
  float gates[128 * 16];     // [slice gate row][batch row]
  float c[16 * 32];          // cell state slice (fp32)
  float h2f[16 * 32];        // fp32 h2 slice for weighted accumulation
  float acc_h[16 * 32];
  float acc_c[16 * 32];
  float halt_part[16 * 16];
  float halt[16];
  float cum[16], rem[16], nup[16], pp[16];
  unsigned done[16];
};

// ---------------- WMMA fragment builders (wave32 layouts per CDNA5 ISA 7.12.2) -
// A: 16x32 bf16 (MxK). lane<16: M=lane, K in {k0..k0+7, k0+16..k0+23};
//                      lane>=16: M=lane-16, K shifted by +8.
__device__ __forceinline__ v16bf frag_A(const bf16* base, int stride, int m, int hf, int k0) {
  const bf16* p0 = base + m * stride + k0 + hf * 8;
  const bf16* p1 = p0 + 16;
  v16bf a;
#pragma unroll
  for (int i = 0; i < 8; ++i) { a[i] = p0[i]; a[8 + i] = p1[i]; }
  return a;
}
// B: 32x16 bf16 (KxN), sourced from row-major weight rows (rows are N columns).
// lane<16: N=lane, K in {k0..k0+7, k0+16..k0+23}; lane>=16: N=lane-16, K shifted by +8.
__device__ __forceinline__ v16bf frag_B(const bf16* base, int stride, int rbase, int n, int hf, int k0) {
  const bf16* p = base + (rbase + n) * stride + k0 + hf * 8;
  v16bf bv;
#pragma unroll
  for (int i = 0; i < 8; ++i) { bv[i] = p[i]; bv[8 + i] = p[16 + i]; }
  return bv;
}

// ---------------- sense-reversing grid barrier (zeroed by prep kernel) --------
__device__ __forceinline__ void grid_barrier(unsigned* bar, unsigned nwg, unsigned& lgen) {
  __syncthreads();
  if (threadIdx.x == 0) {
    __threadfence();
    unsigned g = lgen;
    unsigned prev = atomicAdd(&bar[0], 1u);
    if (prev == nwg - 1u) {
      atomicExch(&bar[0], 0u);
      __threadfence();
      atomicAdd(&bar[1], 1u);
    } else {
      while (atomicAdd(&bar[1], 0u) == g) __builtin_amdgcn_s_sleep(2);
    }
    __threadfence();
    lgen = g + 1u;
  }
  __syncthreads();
}

// ---------------- full-h reload into LDS tile --------------------------------
__device__ __forceinline__ void load_h_tile(Smem& S, const bf16* Hs, int b0, int tid) {
#ifdef USE_ASYNC_LDS
  // 16 rows x 1024 B = 1024 x 16B chunks, direct memory->LDS (ASYNCcnt path)
  for (int i = tid; i < 1024; i += 256) {
    int m = i >> 6, d = i & 63;
    const bf16* src = Hs + (size_t)(b0 + m) * H_ + d * 8;
    bf16* dst = &S.h[m * HSTR + d * 8];
    __builtin_amdgcn_global_load_async_to_lds_b128(
        (v4i_glob*)src, (v4i_lds*)dst, 0, 0);
  }
  __builtin_amdgcn_s_wait_asynccnt(0);
#else
  for (int i = tid; i < 16 * 256; i += 256) {
    int m = i >> 8, d = i & 255;
    reinterpret_cast<unsigned*>(&S.h[m * HSTR])[d] =
        reinterpret_cast<const unsigned*>(Hs + (size_t)(b0 + m) * H_)[d];
  }
#endif
}

// ---------------- prep: fp32 -> bf16 weight staging + barrier init ------------
__global__ void act_prep_kernel(const float* __restrict__ Whh_f, const float* __restrict__ Wih_f,
                                const float* __restrict__ Wout_f,
                                bf16* __restrict__ Whh, bf16* __restrict__ Wih,
                                float* __restrict__ wflag, bf16* __restrict__ Wout,
                                unsigned* __restrict__ bar) {
  const int stride = gridDim.x * blockDim.x;
  const int i0 = blockIdx.x * blockDim.x + threadIdx.x;
  for (int k = i0; k < 2048 * 512; k += stride) Whh[k] = f2bf(Whh_f[k]);
  for (int k = i0; k < 2048 * 64; k += stride) {
    int r = k >> 6, col = k & 63;
    Wih[k] = (col < IN_) ? f2bf(Wih_f[r * (IN_ + 1) + col]) : f2bf(0.0f);
  }
  for (int k = i0; k < 2048; k += stride) wflag[k] = Wih_f[k * (IN_ + 1) + IN_];
  for (int k = i0; k < 80 * 512; k += stride) {
    int r = k >> 9, col = k & 511;
    Wout[k] = (r < OUT_) ? f2bf(Wout_f[r * 512 + col]) : f2bf(0.0f);
  }
  if (i0 < 2) bar[i0] = 0u;
}

// ---------------- persistent ACT-LSTM kernel ---------------------------------
__global__ void __launch_bounds__(256, 1)
act_lstm_kernel(const float* __restrict__ x, const float* __restrict__ bias,
                const float* __restrict__ b_out, const float* __restrict__ W_halt,
                const float* __restrict__ b_halt,
                const bf16* __restrict__ Whh, const bf16* __restrict__ Wih,
                const float* __restrict__ wflag, const bf16* __restrict__ Wout,
                bf16* __restrict__ Hbuf, unsigned* __restrict__ bar,
                float* __restrict__ out, float* __restrict__ pout) {
  extern __shared__ char smem_raw[];
  Smem& S = *reinterpret_cast<Smem*>(smem_raw);

  const int tid  = threadIdx.x;
  const int wave = tid >> 5;
  const int lane = tid & 31;
  const int wg   = blockIdx.x;
  const int bg   = wg >> 4;          // batch group (0..1): rows bg*16..bg*16+15
  const int lid  = wg & 15;          // WGP slot inside batch group
  const int hu0  = lid * 32;         // owned hidden units [hu0, hu0+32)
  const int b0   = bg * 16;
  const int hf   = lane >> 4;        // lane half for WMMA layouts
  const int nn   = lane & 15;        // N column inside 16x16 tile
  const int mrow = lane & 15;        // A-fragment M row

  unsigned lgen = 0;

  // ---- stage this WGP's weight slices into LDS (once; persistent kernel) ----
  for (int i = tid; i < 128 * 256; i += 256) {             // W_hh slice, dwords
    int r = i >> 8, d = i & 255;
    int gr = ((r >> 5) << 9) + hu0 + (r & 31);             // gate*512 + hidden
    reinterpret_cast<unsigned*>(&S.W[r * HSTR])[d] =
        reinterpret_cast<const unsigned*>(Whh + (size_t)gr * 512)[d];
  }
  for (int i = tid; i < 128 * 32; i += 256) {              // W_ih slice
    int r = i >> 5, d = i & 31;
    int gr = ((r >> 5) << 9) + hu0 + (r & 31);
    reinterpret_cast<unsigned*>(&S.Wih[r * 64])[d] =
        reinterpret_cast<const unsigned*>(Wih + (size_t)gr * 64)[d];
  }
  if (lid == 0) {
    for (int i = tid; i < 80 * 256; i += 256) {
      int r = i >> 8, d = i & 255;
      reinterpret_cast<unsigned*>(&S.Wout[r * HSTR])[d] =
          reinterpret_cast<const unsigned*>(Wout + (size_t)r * 512)[d];
    }
  }
  // init state: c=h=0, accumulators zero, publish zero h into buffer 0
  for (int i = tid; i < 512; i += 256) {
    int m = i >> 5, j = i & 31;
    S.c[m * 32 + j] = 0.0f;
    S.acc_h[m * 32 + j] = 0.0f;
    S.acc_c[m * 32 + j] = 0.0f;
    Hbuf[(size_t)(b0 + m) * H_ + hu0 + j] = f2bf(0.0f);
  }
  if (tid < 16) { S.cum[tid] = 0.f; S.rem[tid] = 0.f; S.nup[tid] = 0.f; S.done[tid] = 0u; }
  int cur = 0;
  grid_barrier(bar, NWG, lgen);
  load_h_tile(S, Hbuf + (size_t)cur * (B_ * H_), b0, tid);
  __syncthreads();

  // per-lane constants: bias / flag weight for this lane's gate-row column
  const int rloc = wave * 16 + nn;
  const int grow = ((rloc >> 5) << 9) + hu0 + (rloc & 31);
  const float bias_l  = bias[grow];
  const float wflag_l = wflag[grow];
  const int   ycol    = wave * 16 + nn;
  const float bout_l  = (lid == 0 && wave < 5 && ycol < OUT_) ? b_out[ycol] : 0.0f;
  const float bh      = b_halt[0];

#pragma unroll 1
  for (int t = 0; t < T_; ++t) {
    // ---- load x_t tile (bf16, K padded to 64) ----
    for (int i = tid; i < 16 * 64; i += 256) {
      int m = i >> 6, k = i & 63;
      float v = (k < IN_) ? x[((size_t)(b0 + m) * T_ + t) * IN_ + k] : 0.0f;
      S.xt[i] = f2bf(v);
    }
    if (t + 1 < T_ && tid < 16)
      __builtin_prefetch(&x[((size_t)(b0 + tid) * T_ + (t + 1)) * IN_], 0, 1);
    __syncthreads();

    // ---- gx = x_t @ W_ih_slice^T + bias (2 WMMAs, reused across ponder steps)
    v8f gx;
#pragma unroll
    for (int r = 0; r < 8; ++r) gx[r] = 0.0f;
#pragma unroll
    for (int ks = 0; ks < 2; ++ks) {
      v16bf a  = frag_A(S.xt, 64, mrow, hf, ks * 32);
      v16bf bv = frag_B(S.Wih, 64, wave * 16, nn, hf, ks * 32);
      gx = __builtin_amdgcn_wmma_f32_16x16x32_bf16(false, a, false, bv, (short)0, gx, false, false);
    }
#pragma unroll
    for (int r = 0; r < 8; ++r) gx[r] += bias_l;

    v8f accy;
#pragma unroll
    for (int r = 0; r < 8; ++r) accy[r] = 0.0f;

#pragma unroll 1
    for (int n = 0; n < NPOND; ++n) {
      // ---- gates = gx (+flag at n==0) + h @ W_hh_slice^T ----
      // 4 independent accumulator chains (K split 4x4) to break the serial
      // WMMA D->C dependency chain; issued round-robin for pipeline overlap.
      v8f acc0 = gx, acc1, acc2, acc3;
      if (n == 0) {
#pragma unroll
        for (int r = 0; r < 8; ++r) acc0[r] += wflag_l;    // ACT first-step flag column
      }
#pragma unroll
      for (int r = 0; r < 8; ++r) { acc1[r] = 0.0f; acc2[r] = 0.0f; acc3[r] = 0.0f; }
#pragma unroll
      for (int q = 0; q < 4; ++q) {
        {
          v16bf a  = frag_A(S.h, HSTR, mrow, hf, (0 * 4 + q) * 32);
          v16bf bv = frag_B(S.W, HSTR, wave * 16, nn, hf, (0 * 4 + q) * 32);
          acc0 = __builtin_amdgcn_wmma_f32_16x16x32_bf16(false, a, false, bv, (short)0, acc0, false, false);
        }
        {
          v16bf a  = frag_A(S.h, HSTR, mrow, hf, (1 * 4 + q) * 32);
          v16bf bv = frag_B(S.W, HSTR, wave * 16, nn, hf, (1 * 4 + q) * 32);
          acc1 = __builtin_amdgcn_wmma_f32_16x16x32_bf16(false, a, false, bv, (short)0, acc1, false, false);
        }
        {
          v16bf a  = frag_A(S.h, HSTR, mrow, hf, (2 * 4 + q) * 32);
          v16bf bv = frag_B(S.W, HSTR, wave * 16, nn, hf, (2 * 4 + q) * 32);
          acc2 = __builtin_amdgcn_wmma_f32_16x16x32_bf16(false, a, false, bv, (short)0, acc2, false, false);
        }
        {
          v16bf a  = frag_A(S.h, HSTR, mrow, hf, (3 * 4 + q) * 32);
          v16bf bv = frag_B(S.W, HSTR, wave * 16, nn, hf, (3 * 4 + q) * 32);
          acc3 = __builtin_amdgcn_wmma_f32_16x16x32_bf16(false, a, false, bv, (short)0, acc3, false, false);
        }
      }
#pragma unroll
      for (int r = 0; r < 8; ++r)
        S.gates[(wave * 16 + nn) * 16 + hf * 8 + r] =
            (acc0[r] + acc1[r]) + (acc2[r] + acc3[r]);     // D layout: M = r + 8*half
      __syncthreads();

      // ---- LSTM activations for owned 32 hidden units x 16 batch rows ----
      bf16* Hp = Hbuf + (size_t)(cur ^ 1) * (B_ * H_);
      for (int i = tid; i < 512; i += 256) {
        int m = i >> 5, j = i & 31;
        float gi = S.gates[(j)      * 16 + m];
        float gf = S.gates[(32 + j) * 16 + m];
        float gg = S.gates[(64 + j) * 16 + m];
        float go = S.gates[(96 + j) * 16 + m];
        float c2 = sigf(gf) * S.c[m * 32 + j] + sigf(gi) * tanhf(gg);
        float h2 = sigf(go) * tanhf(c2);
        S.c[m * 32 + j]   = c2;
        S.h2f[m * 32 + j] = h2;
        Hp[(size_t)(b0 + m) * H_ + hu0 + j] = f2bf(h2);
      }
      grid_barrier(bar, NWG, lgen);
      cur ^= 1;
      load_h_tile(S, Hbuf + (size_t)cur * (B_ * H_), b0, tid);
      __syncthreads();

      // ---- halting logistic: deterministic fixed-order reduction (redundant/WG)
      {
        int m = tid >> 4, seg = tid & 15;
        const bf16*  hr = &S.h[m * HSTR + seg * 32];
        const float* wv = W_halt + seg * 32;
        float s = 0.0f;
#pragma unroll 8
        for (int q = 0; q < 32; ++q) s += bf2f(hr[q]) * wv[q];
        S.halt_part[m * 16 + seg] = s;
      }
      __syncthreads();
      if (tid < 16) {
        float s = 0.0f;
#pragma unroll
        for (int q = 0; q < 16; ++q) s += S.halt_part[tid * 16 + q];
        float hl = sigf(s + bh);
        float cum = S.cum[tid];
        unsigned dn = S.done[tid];
        bool is_last = (n == NPOND - 1);
        bool halted  = ((cum + hl) > (1.0f - 0.01f)) || is_last;
        float p = dn ? 0.0f : (halted ? (1.0f - cum) : hl);
        S.rem[tid] += (!dn && halted) ? (1.0f - cum) : 0.0f;
        S.nup[tid] += dn ? 0.0f : 1.0f;
        S.cum[tid]  = cum + (dn ? 0.0f : hl);
        S.done[tid] = dn | (halted ? 1u : 0u);
        S.pp[tid]   = p;
      }
      __syncthreads();

      // ---- weighted state accumulation (fp32) ----
      for (int i = tid; i < 512; i += 256) {
        int m = i >> 5, j = i & 31;
        float p = S.pp[m];
        S.acc_h[m * 32 + j] += p * S.h2f[m * 32 + j];
        S.acc_c[m * 32 + j] += p * S.c[m * 32 + j];
      }
      // ---- output head: y = h2 @ W_out^T (WG lid==0 only, waves 0..4) ----
      if (lid == 0 && wave < 5) {
        v8f y0, y1, y2, y3;
#pragma unroll
        for (int r = 0; r < 8; ++r) { y0[r] = 0.0f; y1[r] = 0.0f; y2[r] = 0.0f; y3[r] = 0.0f; }
#pragma unroll
        for (int q = 0; q < 4; ++q) {
          {
            v16bf a  = frag_A(S.h, HSTR, mrow, hf, (0 * 4 + q) * 32);
            v16bf bv = frag_B(S.Wout, HSTR, wave * 16, nn, hf, (0 * 4 + q) * 32);
            y0 = __builtin_amdgcn_wmma_f32_16x16x32_bf16(false, a, false, bv, (short)0, y0, false, false);
          }
          {
            v16bf a  = frag_A(S.h, HSTR, mrow, hf, (1 * 4 + q) * 32);
            v16bf bv = frag_B(S.Wout, HSTR, wave * 16, nn, hf, (1 * 4 + q) * 32);
            y1 = __builtin_amdgcn_wmma_f32_16x16x32_bf16(false, a, false, bv, (short)0, y1, false, false);
          }
          {
            v16bf a  = frag_A(S.h, HSTR, mrow, hf, (2 * 4 + q) * 32);
            v16bf bv = frag_B(S.Wout, HSTR, wave * 16, nn, hf, (2 * 4 + q) * 32);
            y2 = __builtin_amdgcn_wmma_f32_16x16x32_bf16(false, a, false, bv, (short)0, y2, false, false);
          }
          {
            v16bf a  = frag_A(S.h, HSTR, mrow, hf, (3 * 4 + q) * 32);
            v16bf bv = frag_B(S.Wout, HSTR, wave * 16, nn, hf, (3 * 4 + q) * 32);
            y3 = __builtin_amdgcn_wmma_f32_16x16x32_bf16(false, a, false, bv, (short)0, y3, false, false);
          }
        }
#pragma unroll
        for (int r = 0; r < 8; ++r)
          accy[r] += S.pp[hf * 8 + r] * (((y0[r] + y1[r]) + (y2[r] + y3[r])) + bout_l);
      }
    } // ponder
    __syncthreads();

    // ---- per-timestep outputs ----
    if (lid == 0 && wave < 5 && ycol < OUT_) {
#pragma unroll
      for (int r = 0; r < 8; ++r)
        out[((size_t)(b0 + hf * 8 + r) * T_ + t) * OUT_ + ycol] = accy[r];
    }
    if (tid < 16)
      pout[(size_t)(b0 + tid) * T_ + t] = S.nup[tid] + S.rem[tid];

    // ---- carry: h <- acc_h, c <- acc_c; publish and resync ----
    bf16* Hp = Hbuf + (size_t)(cur ^ 1) * (B_ * H_);
    for (int i = tid; i < 512; i += 256) {
      int m = i >> 5, j = i & 31;
      float wh = S.acc_h[m * 32 + j];
      S.c[m * 32 + j] = S.acc_c[m * 32 + j];
      Hp[(size_t)(b0 + m) * H_ + hu0 + j] = f2bf(wh);
      S.acc_h[m * 32 + j] = 0.0f;
      S.acc_c[m * 32 + j] = 0.0f;
    }
    if (tid < 16) { S.cum[tid] = 0.f; S.rem[tid] = 0.f; S.nup[tid] = 0.f; S.done[tid] = 0u; }
    grid_barrier(bar, NWG, lgen);
    cur ^= 1;
    load_h_tile(S, Hbuf + (size_t)cur * (B_ * H_), b0, tid);
    __syncthreads();
  } // t
}

// ---------------- trailing block softmax: 6 blocks of 11 along features ------
__global__ void act_softmax_kernel(float* __restrict__ out) {
  int i = blockIdx.x * blockDim.x + threadIdx.x;
  if (i >= B_ * T_ * 6) return;
  float* p = out + (size_t)(i / 6) * OUT_ + (i % 6) * 11;
  float mx = p[0];
#pragma unroll
  for (int k = 1; k < 11; ++k) mx = fmaxf(mx, p[k]);
  float e[11], s = 0.0f;
#pragma unroll
  for (int k = 0; k < 11; ++k) { e[k] = __expf(p[k] - mx); s += e[k]; }
  float inv = 1.0f / s;
#pragma unroll
  for (int k = 0; k < 11; ++k) p[k] = e[k] * inv;
}

extern "C" void kernel_launch(void* const* d_in, const int* in_sizes, int n_in,
                              void* d_out, int out_size, void* d_ws, size_t ws_size,
                              hipStream_t stream) {
  (void)in_sizes; (void)n_in; (void)out_size; (void)ws_size;
  const float* x      = (const float*)d_in[0];
  const float* W_ih   = (const float*)d_in[1];
  const float* W_hh   = (const float*)d_in[2];
  const float* b      = (const float*)d_in[3];
  const float* W_out  = (const float*)d_in[4];
  const float* b_out  = (const float*)d_in[5];
  const float* W_halt = (const float*)d_in[6];
  const float* b_halt = (const float*)d_in[7];

  char* ws = (char*)d_ws;
  unsigned* bar   = (unsigned*)ws;                                  // 256 B
  bf16*  Whh_b    = (bf16*) (ws + 256);                             // 2048*512*2
  bf16*  Wih_b    = (bf16*) (ws + 256 + 2097152);                   // 2048*64*2
  float* wflag    = (float*)(ws + 256 + 2097152 + 262144);          // 2048*4
  bf16*  Wout_b   = (bf16*) (ws + 256 + 2097152 + 262144 + 8192);   // 80*512*2
  bf16*  Hbuf     = (bf16*) (ws + 256 + 2097152 + 262144 + 8192 + 81920); // 2*32*512*2

  float* out  = (float*)d_out;
  float* pout = out + (size_t)B_ * T_ * OUT_;

  act_prep_kernel<<<512, 256, 0, stream>>>(W_hh, W_ih, W_out,
                                           Whh_b, Wih_b, wflag, Wout_b, bar);
  act_lstm_kernel<<<NWG, 256, sizeof(Smem), stream>>>(x, b, b_out, W_halt, b_halt,
                                                      Whh_b, Wih_b, wflag, Wout_b,
                                                      Hbuf, bar, out, pout);
  act_softmax_kernel<<<(B_ * T_ * 6 + 255) / 256, 256, 0, stream>>>(out);
}
